// GQA_13537736917615
// MI455X (gfx1250) — compile-verified, matrix operand order checked by reading, not run on hardware
//
#include <hip/hip_runtime.h>
#include <hip/hip_bf16.h>

// ---------------------------------------------------------------------------
// GQA attention block for MI455X (gfx1250)
//   q = x@Wq, k = x@Wk, v = x@Wv  ->  flash attention (GQA group=4)  ->  @Wo
//   B=2, T=2048, D=2048, HQ=32, HKV=8, DH=64
// bf16 WMMA + async-to-LDS double-buffered staging + TDM tile staging.
// ---------------------------------------------------------------------------

typedef __attribute__((ext_vector_type(16))) __bf16 v16bf;
typedef __attribute__((ext_vector_type(8)))  __bf16 v8bf;
typedef __attribute__((ext_vector_type(8)))  float  v8f;
typedef __attribute__((ext_vector_type(4)))  float  v4f;
typedef __attribute__((ext_vector_type(4)))  unsigned int v4u;
typedef __attribute__((ext_vector_type(8)))  int v8i;
typedef __attribute__((ext_vector_type(4)))  int v4i;

#ifndef __has_builtin
#define __has_builtin(x) 0
#endif
#if __has_builtin(__builtin_amdgcn_tensor_load_to_lds)
#define HAVE_TDM 1
#else
#define HAVE_TDM 0
#endif

#define WMMA_BF16(a, b, c) \
  __builtin_amdgcn_wmma_f32_16x16x32_bf16(false, (a), false, (b), (short)0, (c), false, false)

// 16-bit WMMA operand fragment: rows map to lane&15; the lane's 16 K-elements
// are two contiguous 8-element chunks at +0 and +16 elements (p already
// includes the per-half K base: lanes>=16 -> +8 elements).
__device__ __forceinline__ v16bf frag_ld(const __bf16* p) {
  v8bf c0 = *(const v8bf*)(p);
  v8bf c1 = *(const v8bf*)(p + 16);
  return __builtin_shufflevector(c0, c1, 0, 1, 2, 3, 4, 5, 6, 7,
                                 8, 9, 10, 11, 12, 13, 14, 15);
}

__device__ __forceinline__ unsigned lds_off(const void* p) {
  return (unsigned)(unsigned long long)p;  // flat addr[31:0] == LDS offset
}

// VALU-only xor-shuffle within each 16-lane row via v_permlane16_b32.
__device__ __forceinline__ float pl16(float x, int s0, int s1) {
  int xi = __builtin_bit_cast(int, x);
  int r = __builtin_amdgcn_permlane16(xi, xi, s0, s1, false, false);
  return __builtin_bit_cast(float, r);
}
__device__ __forceinline__ float rowmax16(float x) {
  x = fmaxf(x, pl16(x, (int)0x67452301, (int)0xEFCDAB89));  // xor 1
  x = fmaxf(x, pl16(x, (int)0x54761032, (int)0xDCFE98BA));  // xor 2
  x = fmaxf(x, pl16(x, (int)0x32107654, (int)0xBA98FEDC));  // xor 4
  x = fmaxf(x, pl16(x, (int)0xFEDCBA98, (int)0x76543210));  // xor 8
  return x;
}
__device__ __forceinline__ float rowsum16(float x) {
  x += pl16(x, (int)0x67452301, (int)0xEFCDAB89);
  x += pl16(x, (int)0x54761032, (int)0xDCFE98BA);
  x += pl16(x, (int)0x32107654, (int)0xBA98FEDC);
  x += pl16(x, (int)0xFEDCBA98, (int)0x76543210);
  return x;
}

#if HAVE_TDM
// TDM: DMA a 2D bf16 tile [tile1 rows x tile0 cols] (row stride stride0
// elements) from global into LDS, inserting `pada` code DWORD padding every
// `padi` code DWORDs (padded LDS row stride without VALU work).
__device__ __forceinline__ void tdm_load_2d_bf16(unsigned lds_addr,
                                                 const void* gptr, int dim0,
                                                 int dim1, int stride0,
                                                 int tile0, int tile1,
                                                 unsigned padi, unsigned pada) {
  unsigned long long ga = (unsigned long long)gptr;
  v4u g0;
  g0[0] = 1u;  // count=1, user descriptor
  g0[1] = lds_addr;
  g0[2] = (unsigned)ga;
  g0[3] = (unsigned)((ga >> 32) & 0x01FFFFFFull) | (2u << 30);  // type=2
  v8i g1;
  g1[0] = (int)((1u << 16) |                 // data_size = 2 bytes
                (1u << 20) |                 // pad_enable
                (padi << 22) | (pada << 25));
  g1[1] = (int)(((unsigned)dim0 & 0xffffu) << 16);
  g1[2] = (int)((((unsigned)dim0 >> 16) & 0xffffu) |
                (((unsigned)dim1 & 0xffffu) << 16));
  g1[3] = (int)((((unsigned)dim1 >> 16) & 0xffffu) |
                (((unsigned)tile0 & 0xffffu) << 16));
  g1[4] = (int)((unsigned)tile1 & 0xffffu);  // tile_dim1 (tile_dim2 = 0)
  g1[5] = stride0;                           // tensor_dim0_stride[31:0]
  g1[6] = 0;
  g1[7] = 0;
  v4i g2 = {};
  v4i g3 = {};
#if __clang_major__ >= 23
  v8i g4 = {};
  __builtin_amdgcn_tensor_load_to_lds(g0, g1, g2, g3, g4, 0);
#else
  __builtin_amdgcn_tensor_load_to_lds(g0, g1, g2, g3, 0);
#endif
}
#endif

// ---------------------------------------------------------------------------
// GEMM: C[M,N] = A[M,K] * B[K,N]; A f32 or bf16, B f32 weights.
// OUT_MODE: 0 = f32 row-major, 1 = bf16 row-major,
//           2 = bf16 transposed V layout vt[(m>>11)*512 + n][m&2047].
// 128x128 tile, BK=32, double-buffered LDS (1 barrier/iter), 8 waves,
// each wave computes 32x64. bf16 A-tiles staged by the Tensor Data Mover.
// ---------------------------------------------------------------------------
template <int A_BF16, int OUT_MODE>
__global__ __launch_bounds__(256) void gqa_gemm_kernel(
    const void* __restrict__ Av, const float* __restrict__ Bw,
    void* __restrict__ Cv, int M, int N, int K) {
  constexpr int BM = 128, BN = 128, BK = 32;
  constexpr int SA = 40, SB = 40;  // padded strides (80B, 16B-aligned)
  __shared__ __bf16 lA[2][BM * SA];  // [m][k]
  __shared__ __bf16 lB[2][BN * SB];  // transposed: [n][k]

  const int tid  = threadIdx.x;
  const int lane = tid & 31;
  const int w    = tid >> 5;
  const int wm   = w & 3;
  const int wn   = w >> 2;
  const int m0   = blockIdx.y * BM;
  const int n0   = blockIdx.x * BN;

  const int mrow  = lane & 15;
  const int half  = lane >> 4;
  const int kbase = half * 8;

  const float*  Af = (const float*)Av;
  const __bf16* Ab = (const __bf16*)Av;

  const int arow = tid >> 1;        // A loader: 2 threads/row, 16 el each
  const int acol = (tid & 1) * 16;
  const int bk2 = (tid >> 4) * 2;   // B loader: k-pairs packed to b32 stores
  const int bc8 = (tid & 15) * 8;

  auto stageA = [&](int kk, int buf) {
    if (A_BF16) {
#if HAVE_TDM
      if (w == 0)
        tdm_load_2d_bf16(lds_off(&lA[buf][0]),
                         Ab + (size_t)m0 * K + kk, K, M, K, BK, BM,
                         /*padi: 16 DW*/ 3u, /*pada: 4 DW*/ 3u);
#else
      const __bf16* src = Ab + (size_t)(m0 + arow) * K + kk + acol;
      __bf16* dst = &lA[buf][arow * SA + acol];
      *(v8bf*)(dst)     = *(const v8bf*)(src);
      *(v8bf*)(dst + 8) = *(const v8bf*)(src + 8);
#endif
    } else {
      const float* src = Af + (size_t)(m0 + arow) * K + kk + acol;
      if (kk + BK < K) __builtin_prefetch(src + BK, 0, 1);
      v8bf t0, t1;
#pragma unroll
      for (int i = 0; i < 4; i++) {
        v4f f = *(const v4f*)(src + i * 4);
        if (i < 2) {
          t0[i * 4 + 0] = (__bf16)f.x; t0[i * 4 + 1] = (__bf16)f.y;
          t0[i * 4 + 2] = (__bf16)f.z; t0[i * 4 + 3] = (__bf16)f.w;
        } else {
          t1[i * 4 - 8] = (__bf16)f.x; t1[i * 4 - 7] = (__bf16)f.y;
          t1[i * 4 - 6] = (__bf16)f.z; t1[i * 4 - 5] = (__bf16)f.w;
        }
      }
      __bf16* dst = &lA[buf][arow * SA + acol];
      *(v8bf*)(dst)     = t0;
      *(v8bf*)(dst + 8) = t1;
    }
  };

  auto stageB = [&](int kk, int buf) {
    const float* s0 = Bw + (size_t)(kk + bk2) * N + n0 + bc8;
    const float* s1 = s0 + N;
    if (kk + BK < K) __builtin_prefetch(s0 + (size_t)BK * N, 0, 1);
#pragma unroll
    for (int i = 0; i < 8; i += 4) {
      v4f f0 = *(const v4f*)(s0 + i);
      v4f f1 = *(const v4f*)(s1 + i);
#pragma unroll
      for (int j = 0; j < 4; j++) {
        union { __bf16 h[2]; unsigned u; } pk;
        pk.h[0] = (__bf16)f0[j];
        pk.h[1] = (__bf16)f1[j];
        *(unsigned*)&lB[buf][(bc8 + i + j) * SB + bk2] = pk.u;
      }
    }
  };

  v8f acc[2][4] = {};

  stageA(0, 0);
  stageB(0, 0);
#if HAVE_TDM
  if (A_BF16 && w == 0) __builtin_amdgcn_s_wait_tensorcnt(0);
#endif
  __syncthreads();

  for (int kk = 0; kk < K; kk += BK) {
    const int buf = (kk / BK) & 1;
    if (kk + BK < K) {
      stageA(kk + BK, buf ^ 1);
      stageB(kk + BK, buf ^ 1);
    }

    v16bf af[2], bfr[4];
#pragma unroll
    for (int i = 0; i < 2; i++)
      af[i] = frag_ld(&lA[buf][(wm * 32 + i * 16 + mrow) * SA + kbase]);
#pragma unroll
    for (int j = 0; j < 4; j++)
      bfr[j] = frag_ld(&lB[buf][(wn * 64 + j * 16 + mrow) * SB + kbase]);
#pragma unroll
    for (int i = 0; i < 2; i++)
#pragma unroll
      for (int j = 0; j < 4; j++)
        acc[i][j] = WMMA_BF16(af[i], bfr[j], acc[i][j]);

#if HAVE_TDM
    if (A_BF16 && w == 0 && kk + BK < K) __builtin_amdgcn_s_wait_tensorcnt(0);
#endif
    __syncthreads();
  }

#pragma unroll
  for (int i = 0; i < 2; i++)
#pragma unroll
    for (int j = 0; j < 4; j++)
#pragma unroll
      for (int v = 0; v < 8; v++) {
        const int row = m0 + wm * 32 + i * 16 + v + 8 * half;
        const int col = n0 + wn * 64 + j * 16 + mrow;
        const float x = acc[i][j][v];
        if (OUT_MODE == 0) {
          ((float*)Cv)[(size_t)row * N + col] = x;
        } else if (OUT_MODE == 1) {
          ((__bf16*)Cv)[(size_t)row * N + col] = (__bf16)x;
        } else {  // transposed V: row = b*2048 + t, col = kvh*64 + d
          const int bidx = row >> 11;
          const int t = row & 2047;
          ((__bf16*)Cv)[(size_t)(bidx * 512 + col) * 2048 + t] = (__bf16)x;
        }
      }
}

// ---------------------------------------------------------------------------
// Flash attention (non-causal, GQA group=4).
// grid=(T/128, B*HQ); 8 waves, each owns 16 query rows.
// K and (pre-transposed) V tiles staged with global_load_async_to_lds_b128,
// double-buffered: DMA of tile i+1 overlaps WMMA+softmax of tile i.
// Softmax in exp2 domain with raw v_exp_f32 and permlane16 row reductions.
// ---------------------------------------------------------------------------
__global__ __launch_bounds__(256) void gqa_attn_kernel(
    const __bf16* __restrict__ Q, const __bf16* __restrict__ Km,
    const __bf16* __restrict__ Vt, __bf16* __restrict__ O) {
  constexpr int T = 2048, HQ = 32, GROUP_LOG2 = 2;
  constexpr int SK = 72;                // padded stride (144B, 16B-aligned)
  __shared__ __bf16 lK[2][64 * SK];     // [key][d]  (B^T layout for Q*K^T)
  __shared__ __bf16 lV[2][64 * SK];     // [d][key]  (B^T layout for P*V)
  __shared__ __bf16 lP[8 * 16 * SK];    // per-wave P tile [16][SK]

  const int tid   = threadIdx.x;
  const int lane  = tid & 31;
  const int w     = tid >> 5;
  const int mrow  = lane & 15;
  const int half  = lane >> 4;
  const int kbase = half * 8;

  const int bh  = blockIdx.y;
  const int b   = bh / HQ;
  const int h   = bh % HQ;
  const int kvh = h >> GROUP_LOG2;
  const int q0  = blockIdx.x * 128;

  const float sl2 = 0.125f * 1.44269504088896340736f;  // scale * log2(e)

  const __bf16* qrow = Q + (size_t)(b * T + q0 + w * 16 + mrow) * 2048 + h * 64;
  v16bf qa[2];
  qa[0] = frag_ld(qrow + kbase);
  qa[1] = frag_ld(qrow + 32 + kbase);

  v8f o[4] = {};
  float mx[8], lsum[8];
#pragma unroll
  for (int v = 0; v < 8; v++) { mx[v] = -1e30f; lsum[v] = 0.0f; }

  // async loaders: 4 threads per row, 16 bf16 (two b128) each
  const int srow = tid >> 2;        // K: key row / V: d row (0..63)
  const int scol = (tid & 3) * 16;  // column chunk

  const unsigned long long kbase_g =
      (unsigned long long)(Km + (size_t)b * T * 512);
  const unsigned long long vbase_g = (unsigned long long)Vt;
  const int vrow0 = b * 512 + kvh * 64;  // transposed-V row base

  auto stage = [&](int kt, int buf) {
    unsigned ldsa = lds_off(&lK[buf][srow * SK + scol]);
    unsigned goff = (unsigned)(((kt + srow) * 512 + kvh * 64 + scol) * 2);
    asm volatile(
        "global_load_async_to_lds_b128 %0, %1, %2\n\t"
        "global_load_async_to_lds_b128 %0, %1, %2 offset:16"
        :
        : "v"(ldsa), "v"(goff), "s"(kbase_g)
        : "memory");
    unsigned ldsv = lds_off(&lV[buf][srow * SK + scol]);
    unsigned goffv = (unsigned)(((vrow0 + srow) * 2048 + kt + scol) * 2);
    asm volatile(
        "global_load_async_to_lds_b128 %0, %1, %2\n\t"
        "global_load_async_to_lds_b128 %0, %1, %2 offset:16"
        :
        : "v"(ldsv), "v"(goffv), "s"(vbase_g)
        : "memory");
  };

  stage(0, 0);
  asm volatile("s_wait_asynccnt 0x0" ::: "memory");
  __syncthreads();

  for (int kt = 0; kt < T; kt += 64) {
    const int buf = (kt >> 6) & 1;
    if (kt + 64 < T) stage(kt + 64, buf ^ 1);

    // ---- S = Q * K^T (16 q-rows x 64 keys) ----
    v8f s[4] = {};
#pragma unroll
    for (int j = 0; j < 4; j++)
#pragma unroll
      for (int ss = 0; ss < 2; ss++) {
        v16bf bk = frag_ld(&lK[buf][(j * 16 + mrow) * SK + ss * 32 + kbase]);
        s[j] = WMMA_BF16(qa[ss], bk, s[j]);
      }

    // ---- online softmax, exp2 domain (row = v + 8*half) ----
    float mnew[8];
#pragma unroll
    for (int v = 0; v < 8; v++) {
      float tm = -1e30f;
#pragma unroll
      for (int j = 0; j < 4; j++) {
        float sv = s[j][v] * sl2;
        s[j][v] = sv;
        tm = fmaxf(tm, sv);
      }
      mnew[v] = fmaxf(mx[v], rowmax16(tm));
    }
#pragma unroll
    for (int v = 0; v < 8; v++) {
      float rs = 0.0f;
#pragma unroll
      for (int j = 0; j < 4; j++) {
        float p = __builtin_amdgcn_exp2f(s[j][v] - mnew[v]);
        s[j][v] = p;
        rs += p;
      }
      rs = rowsum16(rs);
      const float c = __builtin_amdgcn_exp2f(mx[v] - mnew[v]);
      lsum[v] = lsum[v] * c + rs;
      mx[v] = mnew[v];
#pragma unroll
      for (int d = 0; d < 4; d++) o[d][v] *= c;
    }

    // ---- spill P to this wave's private LDS tile (intra-wave DS ordering
    //      suffices; no workgroup barrier needed) ----
    __bf16* pw = &lP[w * 16 * SK];
#pragma unroll
    for (int j = 0; j < 4; j++)
#pragma unroll
      for (int v = 0; v < 8; v++)
        pw[(v + 8 * half) * SK + j * 16 + mrow] = (__bf16)s[j][v];

    // ---- O += P * V ----
#pragma unroll
    for (int ss = 0; ss < 2; ss++) {
      v16bf pa = frag_ld(&pw[mrow * SK + ss * 32 + kbase]);
#pragma unroll
      for (int d = 0; d < 4; d++) {
        v16bf vb = frag_ld(&lV[buf][(d * 16 + mrow) * SK + ss * 32 + kbase]);
        o[d] = WMMA_BF16(pa, vb, o[d]);
      }
    }

    asm volatile("s_wait_asynccnt 0x0" ::: "memory");
    __syncthreads();  // next tile's async data visible; buffers safe to reuse
  }

  float rl[8];
#pragma unroll
  for (int v = 0; v < 8; v++) rl[v] = __builtin_amdgcn_rcpf(lsum[v]);
#pragma unroll
  for (int d = 0; d < 4; d++)
#pragma unroll
    for (int v = 0; v < 8; v++) {
      const int row = q0 + w * 16 + v + 8 * half;
      const int col = h * 64 + d * 16 + mrow;
      O[(size_t)(b * T + row) * 2048 + col] = (__bf16)(o[d][v] * rl[v]);
    }
}

// ---------------------------------------------------------------------------
extern "C" void kernel_launch(void* const* d_in, const int* in_sizes, int n_in,
                              void* d_out, int out_size, void* d_ws,
                              size_t ws_size, hipStream_t stream) {
  (void)in_sizes; (void)n_in; (void)out_size; (void)ws_size;
  const float* x  = (const float*)d_in[0];
  const float* Wq = (const float*)d_in[1];
  const float* Wk = (const float*)d_in[2];
  const float* Wv = (const float*)d_in[3];
  const float* Wo = (const float*)d_in[4];
  float* out = (float*)d_out;

  const int M = 4096;  // B*T
  const int D = 2048;
  const int KV = 512;

  // workspace: bf16 q (16MB) | k (4MB) | vt transposed (4MB) | attn out (16MB)
  char* ws = (char*)d_ws;
  __bf16* qb = (__bf16*)ws;
  __bf16* kb = (__bf16*)(ws + (size_t)M * D * 2);
  __bf16* vt = (__bf16*)(ws + (size_t)M * D * 2 + (size_t)M * KV * 2);
  __bf16* ab = (__bf16*)(ws + (size_t)M * D * 2 + (size_t)2 * M * KV * 2);

  dim3 blk(256);
  gqa_gemm_kernel<0, 1><<<dim3(D / 128, M / 128), blk, 0, stream>>>(x, Wq, qb, M, D, D);
  gqa_gemm_kernel<0, 1><<<dim3(KV / 128, M / 128), blk, 0, stream>>>(x, Wk, kb, M, KV, D);
  gqa_gemm_kernel<0, 2><<<dim3(KV / 128, M / 128), blk, 0, stream>>>(x, Wv, vt, M, KV, D);
  gqa_attn_kernel<<<dim3(2048 / 128, 64), blk, 0, stream>>>(qb, kb, vt, ab);
  gqa_gemm_kernel<1, 0><<<dim3(D / 128, M / 128), blk, 0, stream>>>(ab, Wo, out, M, D, D);
}
